// SwinBlock_6597069767439
// MI455X (gfx1250) — matrix operand order repaired and checked by reading, not compiled, not run on hardware
//
#include <hip/hip_runtime.h>
#include <hip/hip_bf16.h>

// ---------------------------------------------------------------------------
// Swin block for MI455X (gfx1250, wave32).
//  - All GEMMs via v_wmma_f32_16x16x32_f16 (f32 accumulate).
//  - GEMM B-panels staged to LDS with gfx1250 async copy
//    (global_load_async_to_lds_b128 / s_wait_asynccnt).
// ---------------------------------------------------------------------------

typedef _Float16 h8   __attribute__((ext_vector_type(8)));
typedef _Float16 v16h __attribute__((ext_vector_type(16)));
typedef float    v8f  __attribute__((ext_vector_type(8)));

#define DIM   192
#define HDIM  32
#define NHEAD 6
#define WS    7
#define SS    3
#define IMG   56
#define NTOK  200704      // 64*56*56
#define NWIN  4096        // 64 images * 8*8 windows
#define NPOS  49          // WS*WS

__device__ __forceinline__ v16h zero16() {
    v16h f;
#pragma unroll
    for (int i = 0; i < 16; ++i) f[i] = (_Float16)0;
    return f;
}
__device__ __forceinline__ v8f zero8() {
    v8f f;
#pragma unroll
    for (int i = 0; i < 8; ++i) f[i] = 0.f;
    return f;
}
__device__ __forceinline__ v16h pack2(h8 lo, h8 hi) {
    v16h f;
#pragma unroll
    for (int i = 0; i < 8; ++i) { f[i] = lo[i]; f[i + 8] = hi[i]; }
    return f;
}
__device__ __forceinline__ v8f wmma16(v16h a, v16h b, v8f c) {
    // D = A(16x32 f16) * B(32x16 f16) + C(16x16 f32)
    return __builtin_amdgcn_wmma_f32_16x16x32_f16(false, a, false, b,
                                                  (short)0, c, false, false);
}
// Shifted-window region id along one axis (boundaries at 49 and 53 for 56/7/3)
__device__ __forceinline__ int regid(int g) {
    return (g < IMG - WS) ? 0 : ((g < IMG - SS) ? 1 : 2);
}

// ---------------------------------------------------------------------------
// LayerNorm (+ optional roll(-3,-3) + 7x7 window partition), fp32 -> f16 rows
// ---------------------------------------------------------------------------
template <bool SHIFT>
__global__ __launch_bounds__(256) void k_layernorm(const float* __restrict__ x,
                                                   const float* __restrict__ g,
                                                   const float* __restrict__ b,
                                                   _Float16* __restrict__ out) {
    __shared__ float s1[256];
    __shared__ float s2[256];
    const int t = blockIdx.x;
    const int c = threadIdx.x;
    const float* src;
    long row;
    if (SHIFT) {
        const int bi  = t / (IMG * IMG);
        const int rem = t - bi * IMG * IMG;
        const int sh  = rem / IMG;
        const int sw  = rem - sh * IMG;
        int h = sh + SS; if (h >= IMG) h -= IMG;   // read source of roll(-3)
        int w = sw + SS; if (w >= IMG) w -= IMG;
        src = x + (((long)bi * IMG + h) * IMG + w) * DIM;
        const int wh = sh / WS, pr = sh - wh * WS;
        const int ww = sw / WS, pc = sw - ww * WS;
        row = ((long)bi * 64 + wh * 8 + ww) * NPOS + pr * WS + pc;
    } else {
        src = x + (long)t * DIM;
        row = t;
    }
    const float v = (c < DIM) ? src[c] : 0.f;
    s1[c] = v; s2[c] = v * v;
    __syncthreads();
#pragma unroll
    for (int s = 128; s > 0; s >>= 1) {
        if (c < s) { s1[c] += s1[c + s]; s2[c] += s2[c + s]; }
        __syncthreads();
    }
    const float mean = s1[0] * (1.f / DIM);
    const float var  = s2[0] * (1.f / DIM) - mean * mean;
    const float rstd = rsqrtf(var + 1e-5f);
    if (c < DIM)
        out[row * DIM + c] = (_Float16)((v - mean) * rstd * g[c] + b[c]);
}

// ---------------------------------------------------------------------------
// Cast fp32 W[K,N] -> f16 Wt[N,K] (so B fragments load contiguously in K)
// ---------------------------------------------------------------------------
__global__ __launch_bounds__(256) void k_cast_t(const float* __restrict__ W,
                                                _Float16* __restrict__ Wt,
                                                int K, int N) {
    const long i = (long)blockIdx.x * 256 + threadIdx.x;
    if (i >= (long)K * N) return;
    const int k = (int)(i / N);
    const int n = (int)(i - (long)k * N);
    Wt[(long)n * K + k] = (_Float16)W[i];
}

// ---------------------------------------------------------------------------
// WMMA GEMM: C[M,N] = A[M,K](f16 rowmajor) * Bt[N,K]^T + bias, + epilogue.
// Block tile 256x64: 8 waves x (32x64).  The 64xK B panel is shared by all
// waves and staged into dynamic LDS once per block via the CDNA5 async copy
// path (ASYNCcnt-tracked global_load_async_to_lds_b128).
// ---------------------------------------------------------------------------
enum { EPI_F16 = 0, EPI_GELU = 1, EPI_PROJ = 2, EPI_RES = 3 };

template <int EPI>
__global__ __launch_bounds__(256) void k_gemm(const _Float16* __restrict__ A,
                                              const _Float16* __restrict__ Bt,
                                              const float* __restrict__ bias,
                                              const float* __restrict__ resid,
                                              void* __restrict__ outv,
                                              int M, int N, int K) {
    extern __shared__ __align__(16) char smemraw[];
    const int lane  = threadIdx.x & 31;
    const int wave  = threadIdx.x >> 5;
    const int mlane = lane & 15;
    const int lh    = lane >> 4;
    const int nTiles = N >> 6;
    const int tM = blockIdx.x / nTiles;
    const int tN = blockIdx.x - tM * nTiles;
    const long m0 = (long)tM * 256 + wave * 32;
    const int  n0 = tN * 64;

    // ---- async-stage the shared 64 x K B panel into LDS -------------------
    {
        const int bytes = 64 * K * 2;                  // multiple of 4096
        const char* gsrc = (const char*)(Bt + (long)n0 * K);
        const unsigned ldsBase = __builtin_amdgcn_groupstaticsize();
        for (int off = threadIdx.x * 16; off < bytes; off += 256 * 16) {
            const unsigned lds = ldsBase + (unsigned)off;
            const unsigned long long ga = (unsigned long long)(size_t)(gsrc + off);
            asm volatile("global_load_async_to_lds_b128 %0, %1, off"
                         :: "v"(lds), "v"(ga) : "memory");
        }
        asm volatile("s_wait_asynccnt 0" ::: "memory");
        __syncthreads();
    }
    const _Float16* Bs = (const _Float16*)smemraw;

    const _Float16* a0 = A + (m0 + mlane) * K;
    const _Float16* a1 = A + (m0 + 16 + mlane) * K;
    const _Float16* lb0 = Bs + (long)mlane * K;        // LDS B panel rows
    const _Float16* lb1 = Bs + (long)(16 + mlane) * K;
    const _Float16* lb2 = Bs + (long)(32 + mlane) * K;
    const _Float16* lb3 = Bs + (long)(48 + mlane) * K;

    const int akh = lh * 8;    // A layout: half-wave selects K sub-block of 8
    const int bkh = lh * 8;    // B layout: half-wave selects K sub-block of 8

    v8f acc[2][4];
#pragma unroll
    for (int i = 0; i < 2; ++i)
#pragma unroll
        for (int j = 0; j < 4; ++j) acc[i][j] = zero8();

    for (int k0 = 0; k0 < K; k0 += 32) {
        if (k0 + 32 < K) {                       // gfx1250 global_prefetch_b8
            __builtin_prefetch(a0 + k0 + 32, 0, 0);
            __builtin_prefetch(a1 + k0 + 32, 0, 0);
        }
        const v16h af0 = pack2(*(const h8*)(a0 + k0 + akh),
                               *(const h8*)(a0 + k0 + akh + 16));
        const v16h af1 = pack2(*(const h8*)(a1 + k0 + akh),
                               *(const h8*)(a1 + k0 + akh + 16));
        const v16h bf0 = pack2(*(const h8*)(lb0 + k0 + bkh),
                               *(const h8*)(lb0 + k0 + bkh + 16));
        const v16h bf1 = pack2(*(const h8*)(lb1 + k0 + bkh),
                               *(const h8*)(lb1 + k0 + bkh + 16));
        const v16h bf2 = pack2(*(const h8*)(lb2 + k0 + bkh),
                               *(const h8*)(lb2 + k0 + bkh + 16));
        const v16h bf3 = pack2(*(const h8*)(lb3 + k0 + bkh),
                               *(const h8*)(lb3 + k0 + bkh + 16));
        acc[0][0] = wmma16(af0, bf0, acc[0][0]);
        acc[0][1] = wmma16(af0, bf1, acc[0][1]);
        acc[0][2] = wmma16(af0, bf2, acc[0][2]);
        acc[0][3] = wmma16(af0, bf3, acc[0][3]);
        acc[1][0] = wmma16(af1, bf0, acc[1][0]);
        acc[1][1] = wmma16(af1, bf1, acc[1][1]);
        acc[1][2] = wmma16(af1, bf2, acc[1][2]);
        acc[1][3] = wmma16(af1, bf3, acc[1][3]);
    }

#pragma unroll
    for (int i = 0; i < 2; ++i)
#pragma unroll
        for (int j = 0; j < 4; ++j)
#pragma unroll
            for (int r = 0; r < 8; ++r) {
                const long m = m0 + i * 16 + lh * 8 + r;  // C layout: m = half*8+r
                const int  n = n0 + j * 16 + mlane;
                float v = acc[i][j][r] + bias[n];
                if (EPI == EPI_F16) {
                    ((_Float16*)outv)[m * N + n] = (_Float16)v;
                } else if (EPI == EPI_GELU) {
                    v = 0.5f * v * (1.f + erff(v * 0.70710678118654752f));
                    ((_Float16*)outv)[m * N + n] = (_Float16)v;
                } else if (EPI == EPI_PROJ) {
                    // window-token -> reverse partition -> roll(+3,+3) -> +residual
                    const int mm  = (int)m;
                    const int wn  = mm / NPOS, pos = mm - wn * NPOS;
                    const int bi  = wn >> 6, wl = wn & 63;
                    const int wh  = wl >> 3, ww = wl & 7;
                    const int sh  = wh * WS + pos / WS;
                    const int sw  = ww * WS + pos % WS;
                    int hh = sh + SS; if (hh >= IMG) hh -= IMG;
                    int wv = sw + SS; if (wv >= IMG) wv -= IMG;
                    const long nat = ((long)bi * IMG + hh) * IMG + wv;
                    ((float*)outv)[nat * DIM + n] = resid[nat * DIM + n] + v;
                } else {  // EPI_RES: x2 + mlp_out -> final output
                    ((float*)outv)[m * DIM + n] = resid[m * DIM + n] + v;
                }
            }
}

// ---------------------------------------------------------------------------
// Attention: one wave per (window, head). 49 tokens padded to 64.
// S = q k^T * scale + mask (16 WMMAs), softmax (shfl within 16-lane halves),
// P staged in LDS (C-layout -> A-layout), O = P V (16 WMMAs).
// ---------------------------------------------------------------------------
__global__ __launch_bounds__(32) void k_attn(const _Float16* __restrict__ qkv,
                                             _Float16* __restrict__ out) {
    __shared__ __align__(16) _Float16 P[64 * 64];
    __shared__ __align__(16) _Float16 Vt[HDIM * 64];
    const int lane  = threadIdx.x;
    const int mlane = lane & 15;
    const int lh    = lane >> 4;
    const int win   = blockIdx.x / NHEAD;
    const int head  = blockIdx.x - win * NHEAD;
    const _Float16* base = qkv + (long)win * NPOS * (3 * DIM) + head * HDIM;

    // Stage V transposed: Vt[d][k]  (lane == d, 32 lanes == 32 head dims)
    for (int k = 0; k < 64; ++k) {
        _Float16 v = (_Float16)0;
        if (k < NPOS) v = base[(long)k * (3 * DIM) + 2 * DIM + lane];
        Vt[lane * 64 + k] = v;
    }

    // Q as A-fragments, K as B-fragments (K-dim = head_dim = 32, one WMMA deep)
    v16h qf[4], kf[4];
#pragma unroll
    for (int i = 0; i < 4; ++i) {
        qf[i] = zero16();
        kf[i] = zero16();
        const int m = i * 16 + mlane;
        if (m < NPOS) {
            const _Float16* pq = base + (long)m * (3 * DIM);
            qf[i] = pack2(*(const h8*)(pq + lh * 8),
                          *(const h8*)(pq + lh * 8 + 16));
            const _Float16* pk = pq + DIM;
            kf[i] = pack2(*(const h8*)(pk + lh * 8),
                          *(const h8*)(pk + lh * 8 + 16));
        }
    }

    v8f s[4][4];
#pragma unroll
    for (int mi = 0; mi < 4; ++mi)
#pragma unroll
        for (int nj = 0; nj < 4; ++nj) s[mi][nj] = zero8();
#pragma unroll
    for (int mi = 0; mi < 4; ++mi)
#pragma unroll
        for (int nj = 0; nj < 4; ++nj)
            s[mi][nj] = wmma16(qf[mi], kf[nj], s[mi][nj]);

    // scale + analytic shifted-window mask + row softmax -> P (f16 in LDS)
    const int wi = win & 63, wh = wi >> 3, ww = wi & 7;
    const float scale = 0.17677669529663687f;  // 1/sqrt(32)
#pragma unroll
    for (int mi = 0; mi < 4; ++mi) {
#pragma unroll
        for (int r = 0; r < 8; ++r) {
            const int m = mi * 16 + lh * 8 + r;
            int ridm = 0;
            if (m < NPOS)
                ridm = regid(wh * WS + m / WS) * 3 + regid(ww * WS + m % WS);
            float sv[4];
#pragma unroll
            for (int nj = 0; nj < 4; ++nj) {
                const int n = nj * 16 + mlane;
                float v = s[mi][nj][r] * scale;
                if (n >= NPOS) {
                    v = -1e30f;                       // padded keys: excluded
                } else if (m < NPOS) {
                    const int ridn = regid(wh * WS + n / WS) * 3 +
                                     regid(ww * WS + n % WS);
                    if (ridn != ridm) v -= 100.f;
                }
                sv[nj] = v;
            }
            float mx = fmaxf(fmaxf(sv[0], sv[1]), fmaxf(sv[2], sv[3]));
#pragma unroll
            for (int d = 1; d < 16; d <<= 1)
                mx = fmaxf(mx, __shfl_xor(mx, d, 32));
            float sum = 0.f;
#pragma unroll
            for (int nj = 0; nj < 4; ++nj) {
                sv[nj] = __expf(sv[nj] - mx);
                sum += sv[nj];
            }
#pragma unroll
            for (int d = 1; d < 16; d <<= 1) sum += __shfl_xor(sum, d, 32);
            const float inv = 1.f / sum;
#pragma unroll
            for (int nj = 0; nj < 4; ++nj)
                P[m * 64 + nj * 16 + mlane] = (_Float16)(sv[nj] * inv);
        }
    }
    __syncthreads();  // single-wave WG: lowers to s_nop; LDS is in-order per wave

    // O = P(64x64) * V(64x32): 4x2 accum tiles, 2 K-steps of 32
    v8f o[4][2];
#pragma unroll
    for (int mi = 0; mi < 4; ++mi)
#pragma unroll
        for (int nj = 0; nj < 2; ++nj) o[mi][nj] = zero8();
#pragma unroll
    for (int kj = 0; kj < 2; ++kj) {
        v16h pf[4], vf[2];
#pragma unroll
        for (int mi = 0; mi < 4; ++mi) {
            const _Float16* p = &P[(mi * 16 + mlane) * 64 + kj * 32 + lh * 8];
            pf[mi] = pack2(*(const h8*)p, *(const h8*)(p + 16));
        }
#pragma unroll
        for (int nj = 0; nj < 2; ++nj) {
            const _Float16* p = &Vt[(nj * 16 + mlane) * 64 + kj * 32 + lh * 8];
            vf[nj] = pack2(*(const h8*)p, *(const h8*)(p + 16));
        }
#pragma unroll
        for (int mi = 0; mi < 4; ++mi)
#pragma unroll
            for (int nj = 0; nj < 2; ++nj)
                o[mi][nj] = wmma16(pf[mi], vf[nj], o[mi][nj]);
    }

    _Float16* orow = out + (long)win * NPOS * DIM + head * HDIM;
#pragma unroll
    for (int mi = 0; mi < 4; ++mi)
#pragma unroll
        for (int nj = 0; nj < 2; ++nj)
#pragma unroll
            for (int r = 0; r < 8; ++r) {
                const int m = mi * 16 + lh * 8 + r;
                if (m < NPOS)
                    orow[(long)m * DIM + nj * 16 + mlane] =
                        (_Float16)o[mi][nj][r];
            }
}

// ---------------------------------------------------------------------------
// Host launcher
// ---------------------------------------------------------------------------
extern "C" void kernel_launch(void* const* d_in, const int* in_sizes, int n_in,
                              void* d_out, int out_size, void* d_ws,
                              size_t ws_size, hipStream_t stream) {
    const float* x      = (const float*)d_in[0];
    const float* n1g    = (const float*)d_in[1];
    const float* n1b    = (const float*)d_in[2];
    const float* qkv_w  = (const float*)d_in[3];
    const float* qkv_b  = (const float*)d_in[4];
    const float* proj_w = (const float*)d_in[5];
    const float* proj_b = (const float*)d_in[6];
    const float* n2g    = (const float*)d_in[7];
    const float* n2b    = (const float*)d_in[8];
    const float* w1     = (const float*)d_in[9];
    const float* b1     = (const float*)d_in[10];
    const float* w2     = (const float*)d_in[11];
    const float* b2     = (const float*)d_in[12];
    float* outp = (float*)d_out;

    // Workspace layout (all offsets 16B-aligned); ~540 MB total.
    char* p = (char*)d_ws;
    const size_t SZ_TOKF16 = (size_t)NTOK * DIM * 2;      // 77,070,336
    const size_t SZ_QKV    = (size_t)NTOK * 3 * DIM * 2;  // 231,211,008
    const size_t SZ_X2     = (size_t)NTOK * DIM * 4;      // 154,140,672
    _Float16* xw   = (_Float16*)p; p += SZ_TOKF16;  // LN1'd shifted windows
    _Float16* qkvb = (_Float16*)p; p += SZ_QKV;     // qkv, later reused by MLP hidden
    _Float16* attn = (_Float16*)p; p += SZ_TOKF16;  // attn out (window order)
    float*    x2   = (float*)p;    p += SZ_X2;      // residual stream (natural order)
    _Float16* wqT  = (_Float16*)p; p += (size_t)DIM * 3 * DIM * 2;
    _Float16* wpT  = (_Float16*)p; p += (size_t)DIM * DIM * 2;
    _Float16* w1T  = (_Float16*)p; p += (size_t)DIM * 4 * DIM * 2;
    _Float16* w2T  = (_Float16*)p; p += (size_t)4 * DIM * DIM * 2;
    _Float16* h2   = xw;    // reuse: LN2 output
    _Float16* gbuf = qkvb;  // reuse: qkv+attn regions == NTOK*768*2 bytes exactly

    // Dynamic-LDS B panels: 64*K*2 bytes (24576 for K=192, 98304 for K=768)
    const int LDS_K192 = 64 * 192 * 2;
    const int LDS_K768 = 64 * 768 * 2;
    hipFuncSetAttribute((const void*)k_gemm<EPI_F16>,
                        hipFuncAttributeMaxDynamicSharedMemorySize, LDS_K192);
    hipFuncSetAttribute((const void*)k_gemm<EPI_PROJ>,
                        hipFuncAttributeMaxDynamicSharedMemorySize, LDS_K192);
    hipFuncSetAttribute((const void*)k_gemm<EPI_GELU>,
                        hipFuncAttributeMaxDynamicSharedMemorySize, LDS_K192);
    hipFuncSetAttribute((const void*)k_gemm<EPI_RES>,
                        hipFuncAttributeMaxDynamicSharedMemorySize, LDS_K768);

    // Weight casts/transposes (tiny)
    k_cast_t<<<(DIM * 3 * DIM + 255) / 256, 256, 0, stream>>>(qkv_w, wqT, DIM, 3 * DIM);
    k_cast_t<<<(DIM * DIM + 255) / 256, 256, 0, stream>>>(proj_w, wpT, DIM, DIM);
    k_cast_t<<<(DIM * 4 * DIM + 255) / 256, 256, 0, stream>>>(w1, w1T, DIM, 4 * DIM);
    k_cast_t<<<(4 * DIM * DIM + 255) / 256, 256, 0, stream>>>(w2, w2T, 4 * DIM, DIM);

    // 1) LN1 + roll(-3,-3) + partition
    k_layernorm<true><<<NTOK, 256, 0, stream>>>(x, n1g, n1b, xw);

    // 2) QKV: [200704,192] @ [192,576]   (784 M-blocks x 9 N-tiles)
    k_gemm<EPI_F16><<<(NTOK / 256) * (3 * DIM / 64), 256, LDS_K192, stream>>>(
        xw, wqT, qkv_b, nullptr, qkvb, NTOK, 3 * DIM, DIM);

    // 3) Windowed attention (4096 windows x 6 heads, one wave each)
    k_attn<<<NWIN * NHEAD, 32, 0, stream>>>(qkvb, attn);

    // 4) Proj + reverse-partition + roll(+3,+3) + residual -> x2
    k_gemm<EPI_PROJ><<<(NTOK / 256) * (DIM / 64), 256, LDS_K192, stream>>>(
        attn, wpT, proj_b, x, x2, NTOK, DIM, DIM);

    // 5) LN2
    k_layernorm<false><<<NTOK, 256, 0, stream>>>(x2, n2g, n2b, h2);

    // 6) MLP1 + exact GELU: [200704,192] @ [192,768]
    k_gemm<EPI_GELU><<<(NTOK / 256) * (4 * DIM / 64), 256, LDS_K192, stream>>>(
        h2, w1T, b1, nullptr, gbuf, NTOK, 4 * DIM, DIM);

    // 7) MLP2 + residual: [200704,768] @ [768,192] -> d_out
    k_gemm<EPI_RES><<<(NTOK / 256) * (DIM / 64), 256, LDS_K768, stream>>>(
        gbuf, w2T, b2, x2, outp, NTOK, DIM, 4 * DIM);
}
